// SpeculativeDrafter_82497731821691
// MI455X (gfx1250) — compile-verified, matrix operand order checked by reading, not run on hardware
//
#include <hip/hip_runtime.h>
#include <hip/hip_bf16.h>
#include <stdint.h>

// ---------------------------------------------------------------------------
// Problem constants (from reference setup_inputs)
// ---------------------------------------------------------------------------
#define RB 16            // batch
#define RS 128           // seq len
#define RD 4096          // hidden dim
#define RV 50257         // vocab
#define NUM_DRAFTS 4
#define DRAFT_LENGTH 8
#define NSAMPLES (RB * NUM_DRAFTS * DRAFT_LENGTH)   // 512

typedef __attribute__((ext_vector_type(16))) _Float16 v16h;
typedef __attribute__((ext_vector_type(8)))  float    v8f;

// ---------------------------------------------------------------------------
// Kernel 1: head = softmax(last_hidden @ W_head + b_head)  (reference computes
// then discards this; we compute it with CDNA5 WMMA and park it in d_ws).
// M = 16 (= batch), N = 4 (padded to 16), K = 4096.  One wave (wave32),
// K unrolled x2 -> two v_wmma_f32_16x16x32_f16 per loop body.
// ---------------------------------------------------------------------------
__device__ __forceinline__ void build_frags(
    const float* __restrict__ Arow, const float* __restrict__ W,
    int k0, int khalf, int colC, float colMask, v16h& a, v16h& bf)
{
    #pragma unroll
    for (int vg = 0; vg < 8; ++vg) {
        // 16-bit A-matrix 16x32 VGPR layout (ISA 7.12.2):
        //   vgpr 0-3 : K = 2*vg      + 8*khalf + {0,1}
        //   vgpr 4-7 : K = 16+2*(vg-4) + 8*khalf + {0,1}
        const int kkb = ((vg < 4) ? (2 * vg) : (16 + 2 * (vg - 4))) + 8 * khalf;
        const int k   = k0 + kkb;
        // A: adjacent K-pair is contiguous in memory -> one b64 load.
        const float2 ap = *reinterpret_cast<const float2*>(Arow + k);
        a[2 * vg]     = (_Float16)ap.x;
        a[2 * vg + 1] = (_Float16)ap.y;
        // B: unconditional clamped load + 0/1 mask (branchless, no exec dance).
        bf[2 * vg]     = (_Float16)(W[(size_t)k * NUM_DRAFTS + colC] * colMask);
        bf[2 * vg + 1] = (_Float16)(W[(size_t)(k + 1) * NUM_DRAFTS + colC] * colMask);
    }
}

__global__ __launch_bounds__(32) void head_softmax_wmma(
    const float* __restrict__ hidden,   // (B, S, D)
    const float* __restrict__ W,        // (D, 4) row-major
    const float* __restrict__ bias,     // (4,)
    float* __restrict__ out)            // (16, 4) -> workspace
{
    const int lane  = threadIdx.x;      // 0..31
    const int m     = lane & 15;        // A row (lanes 0-15) / B,N column index
    const int khalf = lane >> 4;        // K-half select per ISA A/B layout

    const float* Arow = hidden + (size_t)m * RS * RD + (size_t)(RS - 1) * RD;
    const int   colC    = (m < NUM_DRAFTS) ? m : 0;
    const float colMask = (m < NUM_DRAFTS) ? 1.0f : 0.0f;

    v8f acc = {};
    for (int k0 = 0; k0 < RD; k0 += 64) {
        v16h a0, b0, a1, b1;
        build_frags(Arow, W, k0,      khalf, colC, colMask, a0, b0);
        build_frags(Arow, W, k0 + 32, khalf, colC, colMask, a1, b1);
        acc = __builtin_amdgcn_wmma_f32_16x16x32_f16(
            false, a0, false, b0, (short)0, acc, false, false);
        acc = __builtin_amdgcn_wmma_f32_16x16x32_f16(
            false, a1, false, b1, (short)0, acc, false, false);
    }

    // C/D layout: lanes 0-15 -> N=lane, M=r; lanes 16-31 -> N=lane-16, M=8+r.
    __shared__ float tile[16][16];
    const float bn = (m < NUM_DRAFTS) ? bias[m] : 0.0f;
    #pragma unroll
    for (int r = 0; r < 8; ++r)
        tile[8 * khalf + r][m] = acc[r] + bn;
    __syncthreads();

    if (lane < 16) {
        const int row = lane;
        float mx = tile[row][0];
        #pragma unroll
        for (int n = 1; n < NUM_DRAFTS; ++n) mx = fmaxf(mx, tile[row][n]);
        float sum = 0.0f;
        float e[NUM_DRAFTS];
        #pragma unroll
        for (int n = 0; n < NUM_DRAFTS; ++n) { e[n] = __expf(tile[row][n] - mx); sum += e[n]; }
        const float inv = 1.0f / sum;
        #pragma unroll
        for (int n = 0; n < NUM_DRAFTS; ++n) out[row * NUM_DRAFTS + n] = e[n] * inv;
    }
}

// ---------------------------------------------------------------------------
// Kernel 2: fused online-softmax + Gumbel-max argmax per (b, draft, pos).
// 512 blocks x 256 threads; each block streams one 50257-elem logits row
// (L2-resident after first touch) exactly once.
// ---------------------------------------------------------------------------
__device__ __forceinline__ uint32_t mix32(uint32_t x) {
    x ^= x >> 16; x *= 0x7feb352dU;
    x ^= x >> 15; x *= 0x846ca68bU;
    x ^= x >> 16;
    return x;
}

#define TPB 256

__global__ __launch_bounds__(TPB) void draft_sample_kernel(
    const float* __restrict__ logits,   // (B, S, V)
    int*   __restrict__ tokens,         // (NSAMPLES,)
    float* __restrict__ probs)          // (NSAMPLES,)
{
    const int sample = blockIdx.x;                  // b*32 + d*8 + p
    const int b      = sample >> 5;
    const float* __restrict__ row =
        logits + ((size_t)b * RS + (RS - 1)) * (size_t)RV;

    const int tid = threadIdx.x;

    // Warm the cold path: kick the first chunk of this row toward L2/L0
    // before the streaming loop (lowers to global_prefetch_b8).
    __builtin_prefetch(row + tid * 16, 0, 0);

    const uint32_t seed = mix32(0x0000002Au ^ (uint32_t)(sample * 0x9E3779B9u));

    float m = -INFINITY, s = 0.0f;                  // online softmax state
    float bestv = -INFINITY, bestraw = 0.0f;        // gumbel-argmax state
    int   besti = 0;

    for (int v = tid; v < RV; v += TPB) {
        const float x = row[v];

        // online softmax (flash-style rescale)
        const float mn = fmaxf(m, x);
        s = s * __expf(m - mn) + __expf(x - mn);
        m = mn;

        // counter-based Gumbel noise for (sample, v)
        const uint32_t h = mix32((uint32_t)v ^ seed);
        const float u = (float)(h >> 8) * 0x1p-24f + 0x1p-25f;   // (0,1)
        const float g = -__logf(-__logf(u));
        const float p = x + g;
        if (p > bestv) { bestv = p; besti = v; bestraw = x; }
    }

    __shared__ float sm[TPB], ss[TPB], sbv[TPB], sbr[TPB];
    __shared__ int   sbi[TPB];
    sm[tid] = m;  ss[tid] = s;
    sbv[tid] = bestv;  sbr[tid] = bestraw;  sbi[tid] = besti;
    __syncthreads();

    for (int off = TPB / 2; off > 0; off >>= 1) {
        if (tid < off) {
            // merge softmax partials
            const float m1 = sm[tid], m2 = sm[tid + off];
            const float mn = fmaxf(m1, m2);
            ss[tid] = ss[tid] * __expf(m1 - mn) + ss[tid + off] * __expf(m2 - mn);
            sm[tid] = mn;
            // merge argmax (tie -> lower index, matching jnp.argmax)
            const float ov = sbv[tid + off];
            if (ov > sbv[tid] || (ov == sbv[tid] && sbi[tid + off] < sbi[tid])) {
                sbv[tid] = ov;
                sbi[tid] = sbi[tid + off];
                sbr[tid] = sbr[tid + off];
            }
        }
        __syncthreads();
    }

    if (tid == 0) {
        tokens[sample] = sbi[0];
        probs[sample]  = __expf(sbr[0] - sm[0]) / ss[0];
    }
}

// ---------------------------------------------------------------------------
// Launch
// ---------------------------------------------------------------------------
extern "C" void kernel_launch(void* const* d_in, const int* in_sizes, int n_in,
                              void* d_out, int out_size, void* d_ws, size_t ws_size,
                              hipStream_t stream) {
    (void)in_sizes; (void)n_in; (void)out_size; (void)ws_size;

    const float* hidden = (const float*)d_in[0];   // (16,128,4096) f32
    const float* logits = (const float*)d_in[1];   // (16,128,50257) f32
    const float* W      = (const float*)d_in[2];   // (4096,4) f32
    const float* bias   = (const float*)d_in[3];   // (4,) f32

    int*   tokens  = (int*)d_out;                  // first 512 outputs (int32)
    float* probsO  = (float*)d_out + NSAMPLES;     // next 512 outputs (f32)
    float* headWs  = (float*)d_ws;                 // discarded head softmax

    hipLaunchKernelGGL(head_softmax_wmma, dim3(1), dim3(32), 0, stream,
                       hidden, W, bias, headWs);
    hipLaunchKernelGGL(draft_sample_kernel, dim3(NSAMPLES), dim3(TPB), 0, stream,
                       logits, tokens, probsO);
}
